// SimpleLSTM_49709951484004
// MI455X (gfx1250) — matrix operand order repaired and checked by reading, not compiled, not run on hardware
//
#include <hip/hip_runtime.h>

typedef __attribute__((ext_vector_type(16))) _Float16 v16h;
typedef __attribute__((ext_vector_type(8)))  float    v8f;

#define H_     64
#define G_     256   // 4*H
#define TOBS_  8
#define PRED_  12
#define WAVES_ 8
#define ROWS_  16    // batch rows per wave
#define BPB_   (WAVES_*ROWS_)

struct Smem {
  _Float16 W0[G_*H_];          // layer0 Whh (enc, then dec)
  _Float16 W1[G_*H_];          // layer1 Wih (enc, then dec)
  _Float16 W2[G_*H_];          // layer1 Whh (enc, then dec)
  float wih0e[G_*2];           // encoder layer0 Wih (D_in=2), kept f32
  float wih0d[G_*2];           // decoder layer0 Wih
  float b0e[G_], b1e[G_], b0d[G_], b1d[G_];  // bih+bhh pre-summed
  float fcw[2*H_];
  float fcbv[4];               // padded to 16B so buffers below stay aligned
  _Float16 h0buf[WAVES_][ROWS_*H_];  // per-wave layer0 hidden (f16)
  _Float16 h1buf[WAVES_][ROWS_*H_];  // per-wave layer1 hidden (f16)
  float    c0buf[WAVES_][ROWS_*H_];  // per-wave layer0 cell state (f32)
  float    c1buf[WAVES_][ROWS_*H_];  // per-wave layer1 cell state (f32)
  float    xbuf[WAVES_][ROWS_*2];    // per-wave 2-wide input / fed-back pred
};

union ABreg { uint4 q[2]; v16h v; };

__device__ __forceinline__ float sig_(float x) {
  float e = __expf(-fabsf(x));
  float r = __builtin_amdgcn_rcpf(1.0f + e);
  return (x >= 0.0f) ? r : e * r;          // 1/(1+e^-x) vs e^x/(1+e^x)
}
__device__ __forceinline__ float tanh_(float x) {
  float e = __expf(-2.0f * fabsf(x));
  float r = (1.0f - e) * __builtin_amdgcn_rcpf(1.0f + e);
  return (x >= 0.0f) ? r : -r;
}

__device__ __forceinline__ v8f wmma16(v16h a, v16h b, v8f c) {
  return __builtin_amdgcn_wmma_f32_16x16x32_f16(false, a, false, b, (short)0, c,
                                                false, false);
}

// A operand: 16x32 f16 tile (M=batch row, K=feature). ISA 7.12.2 layout:
// VGPR 0-3 <- row[kb .. kb+7] (16B), VGPR 4-7 <- row[kb+16 .. kb+23] (16B),
// where arow already includes m*H and the lane-half (+8 for lanes 16-31).
__device__ __forceinline__ void load_A(const _Float16* arow, int kt, ABreg& a) {
  const _Float16* q = arow + kt * 32;
  a.q[0] = *(const uint4*)(q);
  a.q[1] = *(const uint4*)(q + 16);
}

// B operand: 32x16 f16 tile, B(k,n) = W[n][k], W row-major [G][H].
// VGPR 0-3 <- row[kb .. kb+7] (16B), VGPR 4-7 <- row[kb+16 .. kb+23] (16B),
// brow includes n*H and the lane-half (+8 for lanes 16-31); p selects the
// gate tile (stride 4096 elements; folds into the 16-bit DS immediate).
__device__ __forceinline__ void load_B(const _Float16* brow, int p, int kt,
                                       ABreg& b) {
  const _Float16* q = brow + p * (64 * H_) + kt * 32;
  b.q[0] = *(const uint4*)(q);
  b.q[1] = *(const uint4*)(q + 16);
}

// LSTM step, D_in = 2 input (x folded into the accumulator init).
// Gate-column group jg handles ntiles {jg, jg+4, jg+8, jg+12} = (i,f,g,o)
// for hidden columns [16*jg, 16*jg+16). jg loop NOT unrolled (bounded VGPRs).
// B operands are explicitly double-buffered so each WMMA waits only on the
// load issued one tile earlier (partial s_wait_dscnt, latency hidden).
__device__ __forceinline__ void lstm_step_small(
    const float* xb, _Float16* hb, float* cb,
    const _Float16* whh, const float* wih, const float* bias, unsigned lane) {
  unsigned mb = (lane & 16u) ? 8u : 0u;
  unsigned nl = lane & 15u;
  const _Float16* arow = hb + (lane & 15u) * H_ + ((lane & 16u) ? 8u : 0u);
  ABreg a0, a1;
  load_A(arow, 0, a0);
  load_A(arow, 1, a1);
  float xr0[8], xr1[8];
#pragma unroll
  for (int r = 0; r < 8; ++r) {
    xr0[r] = xb[(mb + r) * 2 + 0];
    xr1[r] = xb[(mb + r) * 2 + 1];
  }
#pragma unroll 1
  for (unsigned jg = 0; jg < 4; ++jg) {
    const _Float16* wrow = whh + (jg * 16u + nl) * H_ + ((lane & 16u) ? 8u : 0u);
    v8f acc[4];
#pragma unroll
    for (int p = 0; p < 4; ++p) {
      unsigned n = (jg + 4u * p) * 16u + nl;
      float bsum = bias[n];
      float w0 = wih[n * 2 + 0];
      float w1 = wih[n * 2 + 1];
#pragma unroll
      for (int r = 0; r < 8; ++r)
        acc[p][r] = bsum + xr0[r] * w0 + xr1[r] * w1;
    }
    // 8 WMMAs, tiles s = (p<<1)|kt, B ping-pong one tile ahead
    ABreg bb[2];
    load_B(wrow, 0, 0, bb[0]);
#pragma unroll
    for (int s = 0; s < 8; ++s) {
      if (s < 7) load_B(wrow, (s + 1) >> 1, (s + 1) & 1, bb[(s + 1) & 1]);
      acc[s >> 1] = wmma16(((s & 1) ? a1 : a0).v, bb[s & 1].v, acc[s >> 1]);
    }
#pragma unroll
    for (int r = 0; r < 8; ++r) {
      unsigned idx = (mb + (unsigned)r) * H_ + jg * 16u + nl;
      float iv = sig_(acc[0][r]);
      float fv = sig_(acc[1][r]);
      float gv = tanh_(acc[2][r]);
      float ov = sig_(acc[3][r]);
      float cv = fv * cb[idx] + iv * gv;
      cb[idx] = cv;
      hb[idx] = (_Float16)(ov * tanh_(cv));
    }
  }
}

// LSTM step, 64-dim input (x = previous-layer hidden, in LDS f16).
__device__ __forceinline__ void lstm_step_big(
    const _Float16* xhb, _Float16* hb, float* cb,
    const _Float16* wih, const _Float16* whh, const float* bias, unsigned lane) {
  unsigned mb = (lane & 16u) ? 8u : 0u;
  unsigned nl = lane & 15u;
  unsigned ahalf = (lane & 16u) ? 8u : 0u;
  const _Float16* xrow = xhb + (lane & 15u) * H_ + ahalf;
  const _Float16* hrow = hb + (lane & 15u) * H_ + ahalf;
  ABreg ax0, ax1, ah0, ah1;
  load_A(xrow, 0, ax0);
  load_A(xrow, 1, ax1);
  load_A(hrow, 0, ah0);
  load_A(hrow, 1, ah1);
#pragma unroll 1
  for (unsigned jg = 0; jg < 4; ++jg) {
    unsigned rowoff = (jg * 16u + nl) * H_ + ((lane & 16u) ? 8u : 0u);
    const _Float16* wirow = wih + rowoff;
    const _Float16* whrow = whh + rowoff;
    v8f acc[4];
#pragma unroll
    for (int p = 0; p < 4; ++p) {
      float bsum = bias[(jg + 4u * p) * 16u + nl];
#pragma unroll
      for (int r = 0; r < 8; ++r) acc[p][r] = bsum;
    }
    // 16 WMMAs, tile s: p = s>>2, sub = s&3 -> {Wih k0, Wih k1, Whh k0, Whh k1}
    ABreg bb[2];
    load_B(wirow, 0, 0, bb[0]);
#pragma unroll
    for (int s = 0; s < 16; ++s) {
      if (s < 15) {
        int s2 = s + 1, sub2 = s2 & 3;
        load_B((sub2 < 2) ? wirow : whrow, s2 >> 2, sub2 & 1, bb[s2 & 1]);
      }
      int sub = s & 3;
      v16h av = (sub == 0) ? ax0.v : (sub == 1) ? ax1.v
              : (sub == 2) ? ah0.v : ah1.v;
      acc[s >> 2] = wmma16(av, bb[s & 1].v, acc[s >> 2]);
    }
#pragma unroll
    for (int r = 0; r < 8; ++r) {
      unsigned idx = (mb + (unsigned)r) * H_ + jg * 16u + nl;
      float iv = sig_(acc[0][r]);
      float fv = sig_(acc[1][r]);
      float gv = tanh_(acc[2][r]);
      float ov = sig_(acc[3][r]);
      float cv = fv * cb[idx] + iv * gv;
      cb[idx] = cv;
      hb[idx] = (_Float16)(ov * tanh_(cv));
    }
  }
}

__global__ __launch_bounds__(256)
void lstm_s2s_kernel(
    const float* __restrict__ obs,
    const float* __restrict__ eWih0, const float* __restrict__ eWhh0,
    const float* __restrict__ eBih0, const float* __restrict__ eBhh0,
    const float* __restrict__ eWih1, const float* __restrict__ eWhh1,
    const float* __restrict__ eBih1, const float* __restrict__ eBhh1,
    const float* __restrict__ dWih0, const float* __restrict__ dWhh0,
    const float* __restrict__ dBih0, const float* __restrict__ dBhh0,
    const float* __restrict__ dWih1, const float* __restrict__ dWhh1,
    const float* __restrict__ dBih1, const float* __restrict__ dBhh1,
    const float* __restrict__ fcW, const float* __restrict__ fcb,
    float* __restrict__ out, int B) {
  extern __shared__ char smem_raw[];
  Smem& S = *reinterpret_cast<Smem*>(smem_raw);
  const int tid = threadIdx.x;
  const unsigned lane = (unsigned)tid & 31u;
  const int wave = tid >> 5;
  const int bbase = blockIdx.x * BPB_ + wave * ROWS_;

  // ---- phase 1: encoder weights (f32 -> f16) + all small tensors ----
  for (int i = tid; i < G_ * H_; i += 256) {
    S.W0[i] = (_Float16)eWhh0[i];
    S.W1[i] = (_Float16)eWih1[i];
    S.W2[i] = (_Float16)eWhh1[i];
  }
  for (int i = tid; i < G_ * 2; i += 256) {
    S.wih0e[i] = eWih0[i];
    S.wih0d[i] = dWih0[i];
  }
  for (int i = tid; i < G_; i += 256) {
    S.b0e[i] = eBih0[i] + eBhh0[i];
    S.b1e[i] = eBih1[i] + eBhh1[i];
    S.b0d[i] = dBih0[i] + dBhh0[i];
    S.b1d[i] = dBih1[i] + dBhh1[i];
  }
  for (int i = tid; i < 2 * H_; i += 256) S.fcw[i] = fcW[i];
  if (tid < 2) S.fcbv[tid] = fcb[tid];

  _Float16* h0b = S.h0buf[wave];
  _Float16* h1b = S.h1buf[wave];
  float*    c0b = S.c0buf[wave];
  float*    c1b = S.c1buf[wave];
  float*    xb  = S.xbuf[wave];
  for (int i = (int)lane; i < ROWS_ * H_; i += 32) {
    h0b[i] = (_Float16)0.0f;
    h1b[i] = (_Float16)0.0f;
    c0b[i] = 0.0f;
    c1b[i] = 0.0f;
  }
  __syncthreads();

  // ---- encoder: 8 fused steps (layer0 -> layer1 per step) ----
#pragma unroll 1
  for (int t = 0; t < TOBS_; ++t) {
    if (lane < 16u) {
      int bi = bbase + (int)lane;
      if (bi >= B) bi = B - 1;
      const float* op = obs + (size_t)bi * (TOBS_ * 2) + t * 2;
      xb[lane * 2u + 0u] = op[0];
      xb[lane * 2u + 1u] = op[1];
    }
    lstm_step_small(xb, h0b, c0b, S.W0, S.wih0e, S.b0e, lane);
    lstm_step_big(h0b, h1b, c1b, S.W1, S.W2, S.b1e, lane);
  }

  // ---- swap in decoder weights (reuse the same LDS region) ----
  __syncthreads();
  for (int i = tid; i < G_ * H_; i += 256) {
    S.W0[i] = (_Float16)dWhh0[i];
    S.W1[i] = (_Float16)dWih1[i];
    S.W2[i] = (_Float16)dWhh1[i];
  }
  __syncthreads();

  // decoder x0 = obs[:, -1, :]
  if (lane < 16u) {
    int bi = bbase + (int)lane;
    if (bi >= B) bi = B - 1;
    const float* op = obs + (size_t)bi * (TOBS_ * 2) + (TOBS_ - 1) * 2;
    xb[lane * 2u + 0u] = op[0];
    xb[lane * 2u + 1u] = op[1];
  }

  // ---- decoder: 12 steps; FC head feeds back as next input ----
#pragma unroll 1
  for (int t = 0; t < PRED_; ++t) {
    lstm_step_small(xb, h0b, c0b, S.W0, S.wih0d, S.b0d, lane);
    lstm_step_big(h0b, h1b, c1b, S.W1, S.W2, S.b1d, lane);
    // pred = h1 @ fcW.T + fcb : lane -> (row = lane&15, out dim = lane>>4)
    unsigned m  = lane & 15u;
    unsigned dd = lane >> 4;
    float p = S.fcbv[dd];
    const _Float16* hr = h1b + m * H_;
    const float* fw = S.fcw + dd * H_;
#pragma unroll
    for (int k = 0; k < H_; ++k) p += (float)hr[k] * fw[k];
    int bi = bbase + (int)m;
    if (bi < B) out[(size_t)bi * (PRED_ * 2) + t * 2 + dd] = p;
    xb[m * 2u + dd] = p;   // feed back as next step's x
  }
}

extern "C" void kernel_launch(void* const* d_in, const int* in_sizes, int n_in,
                              void* d_out, int out_size, void* d_ws, size_t ws_size,
                              hipStream_t stream) {
  (void)n_in; (void)out_size; (void)d_ws; (void)ws_size;
  const float* obs   = (const float*)d_in[0];
  const float* eWih0 = (const float*)d_in[1];
  const float* eWhh0 = (const float*)d_in[2];
  const float* eBih0 = (const float*)d_in[3];
  const float* eBhh0 = (const float*)d_in[4];
  const float* eWih1 = (const float*)d_in[5];
  const float* eWhh1 = (const float*)d_in[6];
  const float* eBih1 = (const float*)d_in[7];
  const float* eBhh1 = (const float*)d_in[8];
  const float* dWih0 = (const float*)d_in[9];
  const float* dWhh0 = (const float*)d_in[10];
  const float* dBih0 = (const float*)d_in[11];
  const float* dBhh0 = (const float*)d_in[12];
  const float* dWih1 = (const float*)d_in[13];
  const float* dWhh1 = (const float*)d_in[14];
  const float* dBih1 = (const float*)d_in[15];
  const float* dBhh1 = (const float*)d_in[16];
  const float* fcW   = (const float*)d_in[17];
  const float* fcb   = (const float*)d_in[18];
  float* out = (float*)d_out;

  int B = in_sizes[0] / (TOBS_ * 2);
  int blocks = (B + BPB_ - 1) / BPB_;
  size_t shmem = sizeof(Smem);
  lstm_s2s_kernel<<<blocks, 256, shmem, stream>>>(
      obs, eWih0, eWhh0, eBih0, eBhh0, eWih1, eWhh1, eBih1, eBhh1,
      dWih0, dWhh0, dBih0, dBhh0, dWih1, dWhh1, dBih1, dBhh1,
      fcW, fcb, out, B);
}
